// Block_mamba_77446850282164
// MI455X (gfx1250) — compile-verified, matrix operand order checked by reading, not run on hardware
//
#include <hip/hip_runtime.h>
#include <hip/hip_bf16.h>
#include <math.h>

// ---------------------------------------------------------------------------
// MI455X (gfx1250) fused implementation.
// ~6-8 GFLOP, ~50 MB traffic -> latency/L2-bound; all dense contractions run
// on the WMMA pipe (f16 in, f32 acc, 16x16x32), rfft/irfft are DFT-basis
// GEMMs so the "FFT" is tensor-core work too. GEMM: 16-row A panel staged to
// LDS once (single barrier), then a barrier-free k-loop; each wave computes a
// 16x64 output strip (4 accumulators) reusing every A fragment 4x.
// ---------------------------------------------------------------------------

typedef _Float16 half_t;
typedef __attribute__((ext_vector_type(16))) _Float16 v16h;
typedef __attribute__((ext_vector_type(8)))  float    v8f;

constexpr int B0 = 4, L0 = 576, C0 = 256;
constexpr int DST = 48, DCONV = 4;
constexpr int DIN = 512, DTR = 16, FD = 512;
constexpr int ROWS = B0 * L0;              // 2304
constexpr int NFREQ = 289;                 // rfft bins of 576
constexpr int NFP   = 304;                 // padded to 19*16
constexpr int KCAT  = 608;                 // 2*NFP (xre|xim stacked, 19*32)
constexpr float SCALE = 24.0f;             // sqrt(576)

// ----------------------------- workspace layout ----------------------------
constexpr size_t AL(size_t x) { return (x + 255) & ~size_t(255); }
constexpr size_t o_hA    = 0;                                          // f16 [2304,256]
constexpr size_t o_winT  = AL(o_hA    + (size_t)ROWS * C0   * 2);      // f16 [1024,256]
constexpr size_t o_xr    = AL(o_winT  + (size_t)1024 * C0   * 2);      // f32 [2304,1024]
constexpr size_t o_xc    = AL(o_xr    + (size_t)ROWS * 1024 * 4);      // f32 [2304,512]
constexpr size_t o_xc16  = AL(o_xc    + (size_t)ROWS * DIN  * 4);      // f16 [2304,512]
constexpr size_t o_xpT   = AL(o_xc16  + (size_t)ROWS * DIN  * 2);      // f16 [112,512]
constexpr size_t o_xdbl  = AL(o_xpT   + (size_t)112  * DIN  * 2);      // f32 [2304,112]
constexpr size_t o_delta = AL(o_xdbl  + (size_t)ROWS * 112  * 4);      // f32 [2304,512]
constexpr size_t o_y16   = AL(o_delta + (size_t)ROWS * DIN  * 4);      // f16 [2304,512]
constexpr size_t o_woT   = AL(o_y16   + (size_t)ROWS * DIN  * 2);      // f16 [256,512]
constexpr size_t o_x2    = AL(o_woT   + (size_t)C0   * DIN  * 2);      // f32 [2304,256]
constexpr size_t o_h2    = AL(o_x2    + (size_t)ROWS * C0   * 4);      // f16 [2304,256]
constexpr size_t o_fc1T  = AL(o_h2    + (size_t)ROWS * C0   * 2);      // f16 [512,256]
constexpr size_t o_fT    = AL(o_fc1T  + (size_t)FD   * C0   * 2);      // f16 [512,2304]
constexpr size_t o_fwdB  = AL(o_fT    + (size_t)FD   * ROWS * 2);      // f16 [608,576]
constexpr size_t o_RI    = AL(o_fwdB  + (size_t)KCAT * L0   * 2);      // f16 [4][304,1024]
constexpr size_t o_WcT   = AL(o_RI    + (size_t)B0 * NFP * 1024 * 2);  // f16 [1024,1024]
constexpr size_t o_bcat  = AL(o_WcT   + (size_t)1024 * 1024 * 2);      // f32 [1024]
constexpr size_t o_XcT   = AL(o_bcat  + (size_t)1024 * 4);             // f16 [4][512,608]
constexpr size_t o_ibas  = AL(o_XcT   + (size_t)B0 * FD * KCAT * 2);   // f16 [576,608]
constexpr size_t o_ff16  = AL(o_ibas  + (size_t)L0 * KCAT * 2);        // f16 [2304,512]
constexpr size_t o_fc2T  = AL(o_ff16  + (size_t)ROWS * FD * 2);        // f16 [256,512]

// ------------------------------- GEMM (WMMA) -------------------------------
struct GemmP {
  const half_t* A;  int lda;  long aB;     // row-major f16 A [M,K]
  const half_t* BT; int ldbT; long bB;     // f16 B^T [N,K] (K contiguous)
  int K, N;
  float* Cf;  int ldc;  long cB;
  const float* resid; long rB;
  const float* scale; const float* bias;
  half_t* Ch; int ldch; long chB;
  float oscale;
};

// MODE 0: f32 store            MODE 1: f32 resid+acc
// MODE 2: relu(acc*s+b) -> f16 transposed [col*ldch + row]
// MODE 3: acc*oscale -> RI remap (rows<NFP: real cols 0..511, else imag 512..)
// MODE 4: relu(acc+bias) -> XCAT^T remap ([c][n] / [c][NFP+n])
// MODE 5: f16 row-major        MODE 6: resid + acc*s + b -> f32 (final)
template <int MODE>
__global__ void __launch_bounds__(256) gemm_wmma(GemmP p) {
  extern __shared__ __align__(32) uint32_t lA[];      // 16 rows x K halves
  const int tid  = threadIdx.x;
  const int lane = tid & 31, wv = tid >> 5;
  const int z    = blockIdx.z;
  const long m0  = (long)blockIdx.y * 16;
  const int n0w  = (blockIdx.x * 8 + wv) * 64;        // 4 N-subtiles per wave

  // ---- one-shot stage of the 16xK A panel into LDS (single barrier) ----
  const uint32_t* Ag = reinterpret_cast<const uint32_t*>(p.A + (long)z * p.aB + m0 * p.lda);
  const int ldaD = p.lda >> 1;
  const int KD   = p.K >> 1;                          // dwords per A row
  for (int i = tid; i < (KD << 4); i += 256) {
    const int r = i / KD, c = i - r * KD;
    lA[i] = Ag[(long)r * ldaD + c];
  }
  __syncthreads();

  // ---- 4 B^T column pointers (clamped; stores guarded below) ----
  const half_t* Bbase = p.BT + (long)z * p.bB + ((lane >> 4) * 16);
  const half_t* Bp[4];
#pragma unroll
  for (int s = 0; s < 4; ++s) {
    int nc = n0w + s * 16 + (lane & 15);
    if (nc > p.N - 1) nc = p.N - 1;
    Bp[s] = Bbase + (long)nc * p.ldbT;
  }

  // ---- barrier-free k-loop: 1 A frag (ds_load) feeds 4 WMMAs ----
  v8f acc[4] = {};
  const uint32_t* Arow = &lA[(lane & 15) * KD + ((lane >> 4) * 8)];
  for (int k0 = 0; k0 < p.K; k0 += 32) {
    v16h av = *reinterpret_cast<const v16h*>(Arow);
    Arow += 16;
    if (k0 + 32 < p.K) {
      __builtin_prefetch(Bp[0] + k0 + 32, 0, 1);      // global_prefetch_b8
      __builtin_prefetch(Bp[2] + k0 + 32, 0, 1);
    }
#pragma unroll
    for (int s = 0; s < 4; ++s) {
      v16h bv = *reinterpret_cast<const v16h*>(Bp[s] + k0);
      acc[s] = __builtin_amdgcn_wmma_f32_16x16x32_f16(
          false, av, false, bv, (short)0, acc[s], false, false);
    }
  }

  // ---- fused epilogues ----
#pragma unroll
  for (int s = 0; s < 4; ++s) {
    const int col = n0w + s * 16 + (lane & 15);
    if (col >= p.N) break;
#pragma unroll
    for (int r = 0; r < 8; ++r) {
      const long row = m0 + (lane >> 4) * 8 + r;
      float v = acc[s][r];
      if constexpr (MODE == 0) {
        p.Cf[(long)z * p.cB + row * p.ldc + col] = v;
      } else if constexpr (MODE == 1) {
        p.Cf[(long)z * p.cB + row * p.ldc + col] =
            p.resid[(long)z * p.rB + row * p.ldc + col] + v;
      } else if constexpr (MODE == 2) {
        v = v * p.scale[col] + p.bias[col];
        v = v > 0.f ? v : 0.f;
        p.Ch[(long)col * p.ldch + (long)z * p.chB + row] = (half_t)v;
      } else if constexpr (MODE == 3) {
        half_t* O = p.Ch + (long)z * p.chB;
        v *= p.oscale;
        if (row < NFP) O[row * 1024 + col]               = (half_t)v;
        else           O[(row - NFP) * 1024 + 512 + col] = (half_t)v;
      } else if constexpr (MODE == 4) {
        v += p.bias[col];
        v = v > 0.f ? v : 0.f;
        half_t* O = p.Ch + (long)z * p.chB;
        if (col < 512) O[(long)col * KCAT + row]               = (half_t)v;
        else           O[(long)(col - 512) * KCAT + NFP + row] = (half_t)v;
      } else if constexpr (MODE == 5) {
        p.Ch[(long)z * p.chB + row * p.ldch + col] = (half_t)v;
      } else {  // MODE 6
        p.Cf[row * p.ldc + col] =
            p.resid[row * p.ldc + col] + v * p.scale[col] + p.bias[col];
      }
    }
  }
}

// ------------------------------ aux kernels --------------------------------
__global__ void __launch_bounds__(256) k_ln(const float* __restrict__ X,
    const float* g1, const float* b1, const float* g2, const float* b2,
    half_t* __restrict__ out, int twice) {
  __shared__ float red[256];
  const int t = threadIdx.x;
  const long row = blockIdx.x;
  float v = X[row * C0 + t];
  for (int pass = 0; pass < 1 + twice; ++pass) {
    const float* g = pass ? g2 : g1;
    const float* b = pass ? b2 : b1;
    red[t] = v; __syncthreads();
    for (int s = 128; s > 0; s >>= 1) { if (t < s) red[t] += red[t + s]; __syncthreads(); }
    float m = red[0] * (1.0f / C0); __syncthreads();
    float d = v - m;
    red[t] = d * d; __syncthreads();
    for (int s = 128; s > 0; s >>= 1) { if (t < s) red[t] += red[t + s]; __syncthreads(); }
    float var = red[0] * (1.0f / C0); __syncthreads();
    v = d * rsqrtf(var + 1e-3f) * g[t] + b[t];
  }
  out[row * C0 + t] = (half_t)v;
}

__global__ void k_convT(const float* __restrict__ W, half_t* __restrict__ out,
                        int K, int N) {
  long i = (long)blockIdx.x * blockDim.x + threadIdx.x;
  if (i >= (long)K * N) return;
  int n = (int)(i / K), k = (int)(i % K);
  out[(long)n * K + k] = (half_t)W[(long)k * N + n];
}

__global__ void k_wcat(const float* __restrict__ Wr, const float* __restrict__ Wi,
                       half_t* __restrict__ out) {
  long i = (long)blockIdx.x * blockDim.x + threadIdx.x;
  if (i >= 1024L * 1024) return;
  int j = (int)(i >> 10), k = (int)(i & 1023);
  float v;
  if (j < 512) v = (k < 512) ? Wr[(long)k * 512 + j] : -Wi[(long)(k - 512) * 512 + j];
  else { int j2 = j - 512;
         v = (k < 512) ? Wi[(long)k * 512 + j2] : Wr[(long)(k - 512) * 512 + j2]; }
  out[i] = (half_t)v;
}

__global__ void k_bcat(const float* rb, const float* ib, float* out) {
  int i = blockIdx.x * blockDim.x + threadIdx.x;
  if (i >= 1024) return;
  out[i] = (i < 512) ? rb[i] : ib[i - 512];
}

__global__ void k_fwdB(half_t* __restrict__ out) {   // [KCAT,576] cos / -sin basis
  long i = (long)blockIdx.x * blockDim.x + threadIdx.x;
  if (i >= (long)KCAT * L0) return;
  int r = (int)(i / L0), l = (int)(i % L0);
  float v = 0.f;
  int n = (r < NFP) ? r : r - NFP;
  if (n < NFREQ) {
    float ang = 6.2831853071795864f * (float)((n * l) % L0) / (float)L0;
    v = (r < NFP) ? __cosf(ang) : -__sinf(ang);
  }
  out[i] = (half_t)v;
}

__global__ void k_ibas(half_t* __restrict__ out) {   // [576,KCAT] irfft basis
  long i = (long)blockIdx.x * blockDim.x + threadIdx.x;
  if (i >= (long)L0 * KCAT) return;
  int l = (int)(i / KCAT), k = (int)(i % KCAT);
  float v = 0.f;
  int n = (k < NFP) ? k : k - NFP;
  if (n < NFREQ) {
    float w = (n == 0 || n == NFREQ - 1) ? 1.f : 2.f;       // Hermitian weight
    float ang = 6.2831853071795864f * (float)((n * l) % L0) / (float)L0;
    float c = (1.f / SCALE);                                 // scale/L = 1/24
    v = (k < NFP) ? w * __cosf(ang) * c : -w * __sinf(ang) * c;
  }
  out[i] = (half_t)v;
}

__global__ void k_conv(const float* __restrict__ xr, const float* __restrict__ cw,
                       const float* __restrict__ cb, float* __restrict__ xc,
                       half_t* __restrict__ xc16) {
  long i = (long)blockIdx.x * blockDim.x + threadIdx.x;
  if (i >= (long)ROWS * DIN) return;
  int row = (int)(i / DIN), d = (int)(i % DIN);
  int l = row % L0;
  float acc = cb[d];
#pragma unroll
  for (int k = 0; k < DCONV; ++k) {
    int lk = l - (DCONV - 1) + k;
    if (lk >= 0) acc += xr[(long)(row - (DCONV - 1) + k) * 1024 + d] * cw[k * DIN + d];
  }
  float s = acc / (1.f + __expf(-acc));                      // silu
  xc[i] = s; xc16[i] = (half_t)s;
}

__global__ void k_delta(const float* __restrict__ xdbl, const float* __restrict__ wdt,
                        const float* __restrict__ bdt, float* __restrict__ delta) {
  long i = (long)blockIdx.x * blockDim.x + threadIdx.x;
  if (i >= (long)ROWS * DIN) return;
  int row = (int)(i / DIN), d = (int)(i % DIN);
  const float* dt = xdbl + (long)row * 112;
  float a = bdt[d];
#pragma unroll
  for (int r = 0; r < DTR; ++r) a += dt[r] * wdt[r * DIN + d];
  delta[i] = (a > 20.f) ? a : log1pf(__expf(a));             // softplus
}

__global__ void __launch_bounds__(256) k_scan(const float* __restrict__ delta,
    const float* __restrict__ xc, const float* __restrict__ xdbl,
    const float* __restrict__ xr, const float* __restrict__ Alog,
    const float* __restrict__ Dp, half_t* __restrict__ y16) {
  int gid = blockIdx.x * blockDim.x + threadIdx.x;
  if (gid >= B0 * DIN) return;
  int b = gid / DIN, d = gid % DIN;
  float Av[DST], st[DST];
#pragma unroll
  for (int n = 0; n < DST; ++n) { Av[n] = -__expf(Alog[(long)d * DST + n]); st[n] = 0.f; }
  const float Dd = Dp[d];
  for (int l = 0; l < L0; ++l) {
    long row = (long)b * L0 + l;
    float de = delta[row * DIN + d];
    float u  = xc[row * DIN + d];
    float du = de * u;
    const float* bm = xdbl + row * 112 + DTR;
    const float* cm = bm + DST;
    float y = 0.f;
#pragma unroll
    for (int n = 0; n < DST; ++n) {
      st[n] = st[n] * __expf(de * Av[n]) + du * bm[n];       // h <- e^{dA} h + dBu
      y += st[n] * cm[n];
    }
    y += u * Dd;
    float r = xr[row * 1024 + DIN + d];                      // res gate
    y *= r / (1.f + __expf(-r));
    y16[row * DIN + d] = (half_t)y;
  }
}

// ------------------------------ launcher -----------------------------------
static inline dim3 gr1(long n) { return dim3((unsigned)((n + 255) / 256)); }

extern "C" void kernel_launch(void* const* d_in, const int* in_sizes, int n_in,
                              void* d_out, int out_size, void* d_ws, size_t ws_size,
                              hipStream_t stream) {
  (void)in_sizes; (void)n_in; (void)out_size; (void)ws_size;
  const float* x      = (const float*)d_in[0];
  const float* ln1_g  = (const float*)d_in[1];
  const float* ln1_b  = (const float*)d_in[2];
  const float* mln_g  = (const float*)d_in[3];
  const float* mln_b  = (const float*)d_in[4];
  const float* w_in   = (const float*)d_in[5];
  const float* conv_w = (const float*)d_in[6];
  const float* conv_b = (const float*)d_in[7];
  const float* w_xprj = (const float*)d_in[8];
  const float* w_dt   = (const float*)d_in[9];
  const float* b_dt   = (const float*)d_in[10];
  const float* A_log  = (const float*)d_in[11];
  const float* Dvec   = (const float*)d_in[12];
  const float* w_out  = (const float*)d_in[13];
  const float* ln2_g  = (const float*)d_in[14];
  const float* ln2_b  = (const float*)d_in[15];
  const float* fc1_w  = (const float*)d_in[16];
  const float* bn1_s  = (const float*)d_in[17];
  const float* bn1_b  = (const float*)d_in[18];
  const float* Wr     = (const float*)d_in[19];
  const float* Wi     = (const float*)d_in[20];
  const float* rb     = (const float*)d_in[21];
  const float* ib     = (const float*)d_in[22];
  const float* fc2_w  = (const float*)d_in[23];
  const float* bn2_s  = (const float*)d_in[24];
  const float* bn2_b  = (const float*)d_in[25];

  char* ws = (char*)d_ws;
  half_t* hA   = (half_t*)(ws + o_hA);
  half_t* winT = (half_t*)(ws + o_winT);
  float*  xr   = (float*) (ws + o_xr);
  float*  xc   = (float*) (ws + o_xc);
  half_t* xc16 = (half_t*)(ws + o_xc16);
  half_t* xpT  = (half_t*)(ws + o_xpT);
  float*  xdbl = (float*) (ws + o_xdbl);
  float*  dlt  = (float*) (ws + o_delta);
  half_t* y16  = (half_t*)(ws + o_y16);
  half_t* woT  = (half_t*)(ws + o_woT);
  float*  x2   = (float*) (ws + o_x2);
  half_t* h2   = (half_t*)(ws + o_h2);
  half_t* fc1T = (half_t*)(ws + o_fc1T);
  half_t* fT   = (half_t*)(ws + o_fT);
  half_t* fwdB = (half_t*)(ws + o_fwdB);
  half_t* RI   = (half_t*)(ws + o_RI);
  half_t* WcT  = (half_t*)(ws + o_WcT);
  float*  bcat = (float*) (ws + o_bcat);
  half_t* XcT  = (half_t*)(ws + o_XcT);
  half_t* ibas = (half_t*)(ws + o_ibas);
  half_t* ff16 = (half_t*)(ws + o_ff16);
  half_t* fc2T = (half_t*)(ws + o_fc2T);

  // ---- prep: double-LN, f16 weight transposes, DFT bases, complex weight ---
  k_ln<<<ROWS, 256, 0, stream>>>(x, ln1_g, ln1_b, mln_g, mln_b, hA, 1);
  k_convT<<<gr1((long)256 * 1024), 256, 0, stream>>>(w_in,  winT, 256, 1024);
  k_convT<<<gr1((long)512 * 112),  256, 0, stream>>>(w_xprj, xpT, 512, 112);
  k_convT<<<gr1((long)512 * 256),  256, 0, stream>>>(w_out,  woT, 512, 256);
  k_convT<<<gr1((long)256 * 512),  256, 0, stream>>>(fc1_w, fc1T, 256, 512);
  k_convT<<<gr1((long)512 * 256),  256, 0, stream>>>(fc2_w, fc2T, 512, 256);
  k_wcat<<<gr1(1024L * 1024), 256, 0, stream>>>(Wr, Wi, WcT);
  k_bcat<<<gr1(1024), 256, 0, stream>>>(rb, ib, bcat);
  k_fwdB<<<gr1((long)KCAT * L0), 256, 0, stream>>>(fwdB);
  k_ibas<<<gr1((long)L0 * KCAT), 256, 0, stream>>>(ibas);

  auto shm = [](int K) { return (size_t)16 * K * 2; };

  // ---- G1: xr = hA @ w_in  (2304x256 @ 256x1024) ----
  { GemmP p{}; p.A = hA; p.lda = 256; p.BT = winT; p.ldbT = 256;
    p.K = 256; p.N = 1024; p.Cf = xr; p.ldc = 1024;
    gemm_wmma<0><<<dim3(2, ROWS / 16, 1), 256, shm(256), stream>>>(p); }

  k_conv<<<gr1((long)ROWS * DIN), 256, 0, stream>>>(xr, conv_w, conv_b, xc, xc16);

  // ---- G2: x_dbl = xc @ w_xproj  (2304x512 @ 512x112) ----
  { GemmP p{}; p.A = xc16; p.lda = 512; p.BT = xpT; p.ldbT = 512;
    p.K = 512; p.N = 112; p.Cf = xdbl; p.ldc = 112;
    gemm_wmma<0><<<dim3(1, ROWS / 16, 1), 256, shm(512), stream>>>(p); }

  k_delta<<<gr1((long)ROWS * DIN), 256, 0, stream>>>(xdbl, w_dt, b_dt, dlt);
  k_scan<<<gr1((long)B0 * DIN), 256, 0, stream>>>(dlt, xc, xdbl, xr, A_log, Dvec, y16);

  // ---- G3: x2 = x + y @ w_out ----
  { GemmP p{}; p.A = y16; p.lda = 512; p.BT = woT; p.ldbT = 512;
    p.K = 512; p.N = 256; p.Cf = x2; p.ldc = 256; p.resid = x;
    gemm_wmma<1><<<dim3(1, ROWS / 16, 1), 256, shm(512), stream>>>(p); }

  k_ln<<<ROWS, 256, 0, stream>>>(x2, ln2_g, ln2_b, nullptr, nullptr, h2, 0);

  // ---- G4: f = relu(h2 @ fc1 * s + b) -> f16 transposed [c, b*L+l] ----
  { GemmP p{}; p.A = h2; p.lda = 256; p.BT = fc1T; p.ldbT = 256;
    p.K = 256; p.N = 512; p.scale = bn1_s; p.bias = bn1_b;
    p.Ch = fT; p.ldch = ROWS;
    gemm_wmma<2><<<dim3(1, ROWS / 16, 1), 256, shm(256), stream>>>(p); }

  // ---- G5: rfft as GEMM: [cos;-sin](608x576) @ f[b](576x512) -> RI ----
  { GemmP p{}; p.A = fwdB; p.lda = L0; p.BT = fT; p.ldbT = ROWS; p.bB = L0;
    p.K = L0; p.N = 512; p.Ch = RI; p.chB = (long)NFP * 1024;
    p.oscale = 1.f / SCALE;
    gemm_wmma<3><<<dim3(1, KCAT / 16, B0), 256, shm(L0), stream>>>(p); }

  // ---- G6: complex linear: RI[b](304x1024) @ Wcat(1024x1024) -> XCAT^T ----
  { GemmP p{}; p.A = RI; p.lda = 1024; p.aB = (long)NFP * 1024;
    p.BT = WcT; p.ldbT = 1024;
    p.K = 1024; p.N = 1024; p.bias = bcat;
    p.Ch = XcT; p.chB = (long)FD * KCAT;
    gemm_wmma<4><<<dim3(2, NFP / 16, B0), 256, shm(1024), stream>>>(p); }

  // ---- G7: irfft as GEMM: ibasis(576x608) @ XCAT[b](608x512) -> ff16 ----
  { GemmP p{}; p.A = ibas; p.lda = KCAT; p.BT = XcT; p.ldbT = KCAT;
    p.bB = (long)FD * KCAT;
    p.K = KCAT; p.N = 512; p.Ch = ff16; p.ldch = 512; p.chB = (long)L0 * FD;
    gemm_wmma<5><<<dim3(1, L0 / 16, B0), 256, shm(KCAT), stream>>>(p); }

  // ---- G8: out = x2 + ff @ fc2 * bn2_s + bn2_b ----
  { GemmP p{}; p.A = ff16; p.lda = 512; p.BT = fc2T; p.ldbT = 512;
    p.K = 512; p.N = 256; p.Cf = (float*)d_out; p.ldc = 256;
    p.resid = x2; p.scale = bn2_s; p.bias = bn2_b;
    gemm_wmma<6><<<dim3(1, ROWS / 16, 1), 256, shm(512), stream>>>(p); }
}
